// CategorySpecificLinear_24962349924929
// MI455X (gfx1250) — compile-verified, matrix operand order checked by reading, not run on hardware
//
#include <hip/hip_runtime.h>

// ---------------------------------------------------------------------------
// Per-category affine: y[b] = x[b] @ W[cat[b]] + bias[cat[b]]
// B=64, T=512, I=O=1024, C=32 (fp32).
// Split-bf16 WMMA GEMM (fp32-accurate): f32 = hi(bf16)+lo(bf16);
//   acc += hi_a*hi_b + hi_a*lo_b + lo_a*hi_b  via v_wmma_f32_16x16x32_bf16.
// LDS holds hi/lo planes in fragment order (ds_load_b128 -> v16bf concat).
// Software pipeline: register prefetch of chunk k+1 overlaps WMMA of chunk k;
// LDS double buffer -> one barrier per K iteration.
// Wave tile 32x64 (24 WMMA : 24 ds_load_b128 per chunk); block tile 64x256.
// ---------------------------------------------------------------------------

typedef __attribute__((ext_vector_type(16))) __bf16 v16bf;
typedef __attribute__((ext_vector_type(8)))  __bf16 bf16x8;
typedef __attribute__((ext_vector_type(8)))  float  v8f;
typedef __attribute__((ext_vector_type(4)))  float  f32x4;
typedef __attribute__((ext_vector_type(2)))  unsigned int u32x2;

#define B_DIM 64
#define T_DIM 512
#define I_DIM 1024
#define O_DIM 1024

#define TILE_M 64    // T rows per block
#define TILE_N 256   // O cols per block
#define TILE_K 32

// Per-buffer LDS fragment regions (dword units):
//   A: [m(4)][plane(2)][run(2)][lane(32)][4dw]   = 2048 dw
//   B: [nb(16)][plane(2)][run(2)][lane(32)][4dw] = 8192 dw
#define A_BASE 0
#define B_BASE 2048
#define BUF_DW 10240   // 40 KB per buffer, x2 buffers = 80 KB

static __device__ __forceinline__ unsigned short bfbits(__bf16 b) {
    union { __bf16 b; unsigned short u; } v; v.b = b; return v.u;
}

// split 4 consecutive f32 -> packed hi pairs {h01,h23} and lo pairs {l01,l23}
static __device__ __forceinline__ void split4(f32x4 f, u32x2& hi, u32x2& lo) {
    __bf16 h0 = (__bf16)f[0], h1 = (__bf16)f[1], h2 = (__bf16)f[2], h3 = (__bf16)f[3];
    __bf16 l0 = (__bf16)(f[0] - (float)h0);
    __bf16 l1 = (__bf16)(f[1] - (float)h1);
    __bf16 l2 = (__bf16)(f[2] - (float)h2);
    __bf16 l3 = (__bf16)(f[3] - (float)h3);
    hi[0] = (unsigned int)bfbits(h0) | ((unsigned int)bfbits(h1) << 16);
    hi[1] = (unsigned int)bfbits(h2) | ((unsigned int)bfbits(h3) << 16);
    lo[0] = (unsigned int)bfbits(l0) | ((unsigned int)bfbits(l1) << 16);
    lo[1] = (unsigned int)bfbits(l2) | ((unsigned int)bfbits(l3) << 16);
}

__global__ __launch_bounds__(256)
void CategorySpecificLinear_kernel(const float* __restrict__ x,
                                   const int*   __restrict__ cat,
                                   const float* __restrict__ W,
                                   const float* __restrict__ bias,
                                   float*       __restrict__ out) {
    __shared__ __align__(16) unsigned int lds[2 * BUF_DW];

    const int b  = blockIdx.z;
    const int t0 = blockIdx.y * TILE_M;
    const int o0 = blockIdx.x * TILE_N;
    const int c  = cat[b];

    const float* xb = x + ((size_t)b * T_DIM + t0) * I_DIM;
    const float* Wc = W + (size_t)c * I_DIM * O_DIM + o0;

    const int tid  = threadIdx.x;
    const int lane = tid & 31;
    const int wave = tid >> 5;
    const int wm   = wave & 1;   // wave's 32-row M half (2 x 16-row blocks)
    const int ng   = wave >> 1;  // wave's 64-col N group (4 x 16-col blocks)

    // ---- loop-invariant staging geometry ----
    // x: 512 quads (64 rows x 8 quads), 2 per thread
    int xoff[2], xdst[2];
#pragma unroll
    for (int j = 0; j < 2; ++j) {
        int e = j * 256 + tid, row = e >> 3, kq = e & 7;  // k = 4*kq
        xoff[j] = row * I_DIM + 4 * kq;
        int mm = row >> 4, fr = row & 15;
        int half = (kq >> 1) & 1, run = kq >> 2, dws = (kq & 1) * 2;
        xdst[j] = A_BASE + ((mm * 2 + 0) * 2 + run) * 128 + (fr + 16 * half) * 4 + dws;
    }
    // W: 2048 quads (256 cols x 8 K-quads), 8 per thread
    int woff[8], wdst[8];
#pragma unroll
    for (int j = 0; j < 8; ++j) {
        int e = j * 256 + tid, col = e & 255, kq = e >> 8;  // k = 4*kq
        woff[j] = 4 * kq * O_DIM + col;
        int nb = col >> 4, fc = col & 15;
        int half = (kq >> 1) & 1, run = kq >> 2, dws = (kq & 1) * 2;
        wdst[j] = B_BASE + ((nb * 2 + 0) * 2 + run) * 128 + (fc + 16 * half) * 4 + dws;
    }

    v8f acc[2][4];
#pragma unroll
    for (int mi = 0; mi < 2; ++mi)
#pragma unroll
        for (int n = 0; n < 4; ++n) acc[mi][n] = (v8f){};

    f32x4 xr[2], wr[8];

    auto load_chunk = [&](int kc) {
#pragma unroll
        for (int j = 0; j < 2; ++j)
            xr[j] = *(const f32x4*)&xb[xoff[j] + kc];
        const float* Wck = Wc + (size_t)kc * O_DIM;
#pragma unroll
        for (int j = 0; j < 8; ++j) {
            f32x4 f;
            f[0] = Wck[woff[j]];
            f[1] = Wck[woff[j] + O_DIM];
            f[2] = Wck[woff[j] + 2 * O_DIM];
            f[3] = Wck[woff[j] + 3 * O_DIM];
            wr[j] = f;
        }
    };
    auto store_chunk = [&](int buf) {
        unsigned int* L = lds + buf * BUF_DW;
#pragma unroll
        for (int j = 0; j < 2; ++j) {
            u32x2 hi, lo; split4(xr[j], hi, lo);
            *(u32x2*)&L[xdst[j]]       = hi;   // hi plane
            *(u32x2*)&L[xdst[j] + 256] = lo;   // lo plane (plane stride = 256 dw)
        }
#pragma unroll
        for (int j = 0; j < 8; ++j) {
            u32x2 hi, lo; split4(wr[j], hi, lo);
            *(u32x2*)&L[wdst[j]]       = hi;
            *(u32x2*)&L[wdst[j] + 256] = lo;
        }
    };

    // ---- pipeline prologue: chunk 0 into buffer 0 ----
    load_chunk(0);
    store_chunk(0);
    __syncthreads();

    // ---- main pipeline: one barrier per iteration ----
    for (int i = 0; i < I_DIM / TILE_K; ++i) {
        if (i < I_DIM / TILE_K - 1)
            load_chunk((i + 1) * TILE_K);   // global loads overlap WMMAs below

        const unsigned int* L = lds + (i & 1) * BUF_DW;

        // A fragments for both 16-row blocks of this wave's 32-row half
        v16bf Ahi[2], Alo[2];
#pragma unroll
        for (int mi = 0; mi < 2; ++mi) {
            const unsigned int* Ab =
                L + A_BASE + ((wm * 2 + mi) * 4) * 128 + lane * 4;
            bf16x8 ah0 = *(const bf16x8*)(Ab);
            bf16x8 ah1 = *(const bf16x8*)(Ab + 128);
            bf16x8 al0 = *(const bf16x8*)(Ab + 256);
            bf16x8 al1 = *(const bf16x8*)(Ab + 384);
            Ahi[mi] = __builtin_shufflevector(ah0, ah1,
                          0,1,2,3,4,5,6,7,8,9,10,11,12,13,14,15);
            Alo[mi] = __builtin_shufflevector(al0, al1,
                          0,1,2,3,4,5,6,7,8,9,10,11,12,13,14,15);
        }

#pragma unroll
        for (int n = 0; n < 4; ++n) {
            const unsigned int* Bb =
                L + B_BASE + ((ng * 4 + n) * 4) * 128 + lane * 4;
            bf16x8 bh0 = *(const bf16x8*)(Bb);
            bf16x8 bh1 = *(const bf16x8*)(Bb + 128);
            bf16x8 bl0 = *(const bf16x8*)(Bb + 256);
            bf16x8 bl1 = *(const bf16x8*)(Bb + 384);
            v16bf bhi = __builtin_shufflevector(bh0, bh1,
                            0,1,2,3,4,5,6,7,8,9,10,11,12,13,14,15);
            v16bf blo = __builtin_shufflevector(bl0, bl1,
                            0,1,2,3,4,5,6,7,8,9,10,11,12,13,14,15);

#pragma unroll
            for (int mi = 0; mi < 2; ++mi) {
                v8f a = acc[mi][n];
                a = __builtin_amdgcn_wmma_f32_16x16x32_bf16(false, Ahi[mi], false, blo,
                                                            (short)0, a, false, false);
                a = __builtin_amdgcn_wmma_f32_16x16x32_bf16(false, Alo[mi], false, bhi,
                                                            (short)0, a, false, false);
                a = __builtin_amdgcn_wmma_f32_16x16x32_bf16(false, Ahi[mi], false, bhi,
                                                            (short)0, a, false, false);
                acc[mi][n] = a;
            }
        }

        if (i < I_DIM / TILE_K - 1)
            store_chunk((i + 1) & 1);       // other buffer: no race with readers
        __syncthreads();
    }

    // ---- epilogue: bias + store ----
    // C/D layout: VGPR r, lanes 0-15 -> M=r, lanes 16-31 -> M=r+8; N = lane%16.
    const int rbase = (lane >> 4) * 8;
    const int ncol  = lane & 15;
    float* ob = out + ((size_t)b * T_DIM + t0) * O_DIM + o0;
#pragma unroll
    for (int mi = 0; mi < 2; ++mi) {
        const int row0 = wm * 32 + mi * 16 + rbase;
#pragma unroll
        for (int n = 0; n < 4; ++n) {
            const int col = ng * 64 + n * 16 + ncol;
            const float bv = bias[c * O_DIM + o0 + col];
            v8f a = acc[mi][n];
#pragma unroll
            for (int r = 0; r < 8; ++r) {
                ob[(size_t)(row0 + r) * O_DIM + col] = a[r] + bv;
            }
        }
    }
}

extern "C" void kernel_launch(void* const* d_in, const int* in_sizes, int n_in,
                              void* d_out, int out_size, void* d_ws, size_t ws_size,
                              hipStream_t stream) {
    const float* x    = (const float*)d_in[0];
    const int*   cat  = (const int*)d_in[1];
    const float* W    = (const float*)d_in[2];
    const float* bias = (const float*)d_in[3];
    float*       out  = (float*)d_out;

    dim3 grid(O_DIM / TILE_N, T_DIM / TILE_M, B_DIM);
    dim3 block(256);
    CategorySpecificLinear_kernel<<<grid, block, 0, stream>>>(x, cat, W, bias, out);
}
